// Attention_17532056502607
// MI455X (gfx1250) — matrix operand order repaired and checked by reading, not compiled, not run on hardware
//
#include <hip/hip_runtime.h>
#include <hip/hip_bf16.h>

typedef __attribute__((ext_vector_type(16))) _Float16 v16h;
typedef __attribute__((ext_vector_type(8)))  _Float16 v8h;
typedef __attribute__((ext_vector_type(4)))  _Float16 v4h;
typedef __attribute__((ext_vector_type(8)))  float    v8f;

#define BATCH 8
#define NH    8
#define SEQ   2304
#define CDIM  384
#define HD    48
#define HDP   64

// f16 A-fragment (16x32): one contiguous 16-half run at +2*kb2,
// where kb2 = 0 (lanes 0-15) or 8 (lanes 16-31).
static __device__ __forceinline__ v16h load_a_frag(const _Float16* rowptr, int kb2) {
  v8h lo = *(const v8h*)(rowptr + 2 * kb2);
  v8h hi = *(const v8h*)(rowptr + 2 * kb2 + 8);
  return __builtin_shufflevector(lo, hi, 0, 1, 2, 3, 4, 5, 6, 7,
                                         8, 9, 10, 11, 12, 13, 14, 15);
}

static __device__ __forceinline__ v8f wmma_f16(v16h a, v16h b, v8f c) {
  return __builtin_amdgcn_wmma_f32_16x16x32_f16(false, a, false, b, (short)0, c,
                                                false, false);
}

// ---------------------------------------------------------------------------
// Kernel 0: f32 -> f16 conversion (x and weights, done once).
// ---------------------------------------------------------------------------
__global__ void cvt_kernel(const float* __restrict__ src, _Float16* __restrict__ dst,
                           int n4) {
  int i = blockIdx.x * blockDim.x + threadIdx.x;
  if (i < n4) {
    float4 v = ((const float4*)src)[i];
    v4h h;
    h[0] = (_Float16)v.x; h[1] = (_Float16)v.y;
    h[2] = (_Float16)v.z; h[3] = (_Float16)v.w;
    ((v4h*)dst)[i] = h;
  }
}

// ---------------------------------------------------------------------------
// Kernel 1: fused QKV projection, all-f16 operands, one-step software pipeline.
// wh rows 0..383 = q_w, 384..1151 = kv_w (k: 384..767, v: 768..1151).
// Grid (18432/64, 18), 128 threads (4 waves x 16 rows x 64 cols).
// ---------------------------------------------------------------------------
__global__ void qkv_kernel(const _Float16* __restrict__ xh,
                           const _Float16* __restrict__ wh,
                           const float* __restrict__ q_b,
                           const float* __restrict__ kv_b,
                           _Float16* __restrict__ qp,   // [B,nh,N,64] row-major
                           _Float16* __restrict__ kp,   // [B,nh,N,64] row-major
                           _Float16* __restrict__ vT)   // [B,nh,48,N] transposed
{
  const int lane  = threadIdx.x & 31;
  const int wave  = threadIdx.x >> 5;
  const int m0    = blockIdx.x * 64 + wave * 16;
  const int f0    = blockIdx.y * 64;
  const int b     = m0 / SEQ;
  const int nbase = m0 % SEQ;
  const int kb2   = (lane < 16) ? 0 : 8;
  const int koff  = (lane < 16) ? 0 : 16;

  const _Float16* arow = xh + (size_t)(m0 + (lane & 15)) * CDIM;

  int ofeat[4];
  const _Float16* brow[4];
#pragma unroll
  for (int nt = 0; nt < 4; ++nt) {
    ofeat[nt] = f0 + nt * 16 + (lane & 15);
    brow[nt]  = wh + (size_t)ofeat[nt] * CDIM + koff;
  }

  // Prologue: fragments for kt = 0.
  v16h a = load_a_frag(arow, kb2);
  v16h bf[4];
#pragma unroll
  for (int nt = 0; nt < 4; ++nt) bf[nt] = *(const v16h*)(brow[nt]);

  v8f acc[4] = {};
#pragma unroll 2
  for (int kt = 0; kt < 12; ++kt) {
    // Issue next iteration's loads before consuming this iteration's
    // fragments so the WMMAs don't wait on loadcnt==0.
    const int ktn = (kt < 11) ? kt + 1 : kt;
    v16h an = load_a_frag(arow + ktn * 32, kb2);
    v16h bn[4];
#pragma unroll
    for (int nt = 0; nt < 4; ++nt) bn[nt] = *(const v16h*)(brow[nt] + ktn * 32);

#pragma unroll
    for (int nt = 0; nt < 4; ++nt) acc[nt] = wmma_f16(a, bf[nt], acc[nt]);

    a = an;
#pragma unroll
    for (int nt = 0; nt < 4; ++nt) bf[nt] = bn[nt];
  }

#pragma unroll
  for (int nt = 0; nt < 4; ++nt) {
    const int o = ofeat[nt];
    const float bias = (o < 384) ? q_b[o] : kv_b[o - 384];
#pragma unroll
    for (int r = 0; r < 8; ++r) {
      int row = (lane < 16) ? r : 8 + r;
      int n   = nbase + row;
      float val = acc[nt][r] + bias;
      if (o < 384) {
        int h = o / HD, d = o % HD;
        qp[(((size_t)(b * NH + h) * SEQ) + n) * HDP + d] = (_Float16)val;
      } else if (o < 768) {
        int ok = o - 384; int h = ok / HD, d = ok % HD;
        kp[(((size_t)(b * NH + h) * SEQ) + n) * HDP + d] = (_Float16)val;
      } else {
        int ov = o - 768; int h = ov / HD, d = ov % HD;
        vT[(((size_t)(b * NH + h) * HD) + d) * SEQ + n] = (_Float16)val;
      }
    }
  }
}

// ---------------------------------------------------------------------------
// Kernel 2: streaming (flash) attention.  Grid (B*nh, N/64), 128 threads.
// K row-major [n][64], V transposed [48][N]: every B-fragment is one
// contiguous 32-byte run per lane.  Per iteration: issue all K loads, then
// all V loads, S-WMMAs wait only for K; softmax overlaps V loads.
// ---------------------------------------------------------------------------
__global__ void attn_kernel(const _Float16* __restrict__ qp,
                            const _Float16* __restrict__ kp,
                            const _Float16* __restrict__ vT,
                            _Float16* __restrict__ att) // [B,N,384]
{
  __shared__ _Float16 plds[4][16][40];  // stride 40 halfs: 16B-aligned, no bank conflicts

  const int lane = threadIdx.x & 31;
  const int wave = threadIdx.x >> 5;
  const int bh   = blockIdx.x;
  const int n0   = blockIdx.y * 64 + wave * 16;
  const int b    = bh >> 3, h = bh & 7;
  const int kb2  = (lane < 16) ? 0 : 8;
  const int koff = (lane < 16) ? 0 : 16;
  const int col  = lane & 15;

  const _Float16* qbase = qp + (size_t)bh * SEQ * HDP;
  const _Float16* kbase = kp + (size_t)bh * SEQ * HDP;
  const _Float16* vbase = vT + (size_t)bh * HD * SEQ;

  v16h qa[2];
  {
    const _Float16* qrow = qbase + (size_t)(n0 + col) * HDP;
    qa[0] = load_a_frag(qrow, kb2);
    qa[1] = load_a_frag(qrow + 32, kb2);
  }

  v8f o0 = {}, o1 = {}, o2 = {};
  float mr[8], lr[8];
#pragma unroll
  for (int r = 0; r < 8; ++r) { mr[r] = -1e30f; lr[r] = 0.0f; }

  const float scale = 0.14433756729740643f;   // 1/sqrt(48)

  for (int kb = 0; kb < SEQ / 32; ++kb) {
    const int j0 = kb * 32;
    const _Float16* krow0 = kbase + (size_t)(j0 + col) * HDP + koff;
    const _Float16* krow1 = krow0 + (size_t)16 * HDP;

    // Issue all K-fragment loads first...
    v16h kf00 = *(const v16h*)(krow0);
    v16h kf01 = *(const v16h*)(krow0 + 32);
    v16h kf10 = *(const v16h*)(krow1);
    v16h kf11 = *(const v16h*)(krow1 + 32);
    // ...then all V-fragment loads (consumed only after softmax).
    v16h vf0 = *(const v16h*)(vbase + (size_t)(0 * 16 + col) * SEQ + j0 + koff);
    v16h vf1 = *(const v16h*)(vbase + (size_t)(1 * 16 + col) * SEQ + j0 + koff);
    v16h vf2 = *(const v16h*)(vbase + (size_t)(2 * 16 + col) * SEQ + j0 + koff);

    // S = Q @ K^T : two 16x16 f32 tiles
    v8f s0 = {}, s1 = {};
    s0 = wmma_f16(qa[0], kf00, s0);
    s1 = wmma_f16(qa[0], kf10, s1);
    s0 = wmma_f16(qa[1], kf01, s0);
    s1 = wmma_f16(qa[1], kf11, s1);

    // Online softmax; row reductions are 16-lane butterflies within each half.
#pragma unroll
    for (int r = 0; r < 8; ++r) {
      float v0 = s0[r] * scale;
      float v1 = s1[r] * scale;
      float t  = fmaxf(v0, v1);
      t = fmaxf(t, __shfl_xor(t, 1, 16));
      t = fmaxf(t, __shfl_xor(t, 2, 16));
      t = fmaxf(t, __shfl_xor(t, 4, 16));
      t = fmaxf(t, __shfl_xor(t, 8, 16));
      float mnew  = fmaxf(mr[r], t);
      float alpha = __expf(mr[r] - mnew);
      float p0 = __expf(v0 - mnew);
      float p1 = __expf(v1 - mnew);
      float rs = p0 + p1;
      rs += __shfl_xor(rs, 1, 16);
      rs += __shfl_xor(rs, 2, 16);
      rs += __shfl_xor(rs, 4, 16);
      rs += __shfl_xor(rs, 8, 16);
      lr[r] = lr[r] * alpha + rs;
      mr[r] = mnew;
      o0[r] = o0[r] * alpha;
      o1[r] = o1[r] * alpha;
      o2[r] = o2[r] * alpha;
      int row = (lane < 16) ? r : 8 + r;
      plds[wave][row][col]      = (_Float16)p0;
      plds[wave][row][col + 16] = (_Float16)p1;
    }
    // Per-wave buffer: same-wave DS ops are processed in order; fence the
    // store->load hazard with the CDNA5 split DS counter.
    asm volatile("s_wait_dscnt 0" ::: "memory");

    v16h pa = load_a_frag(&plds[wave][col][0], kb2);

    // O += P @ V  (K = 32 keys, 3 column tiles over hd = 48)
    o0 = wmma_f16(pa, vf0, o0);
    o1 = wmma_f16(pa, vf1, o1);
    o2 = wmma_f16(pa, vf2, o2);
  }

  // Normalize; emit f16 rows laid out [B, N, C] for the projection GEMM.
#pragma unroll
  for (int r = 0; r < 8; ++r) {
    int row = (lane < 16) ? r : 8 + r;
    float invl = 1.0f / lr[r];
    size_t base = ((size_t)(b * SEQ + n0 + row)) * CDIM + h * HD;
    att[base + 0  + col] = (_Float16)(o0[r] * invl);
    att[base + 16 + col] = (_Float16)(o1[r] * invl);
    att[base + 32 + col] = (_Float16)(o2[r] * invl);
  }
}

// ---------------------------------------------------------------------------
// Kernel 3: output projection.  out = att @ proj_w^T + proj_b (f32 out),
// same one-step software pipeline as kernel 1.
// ---------------------------------------------------------------------------
__global__ void proj_kernel(const _Float16* __restrict__ att,
                            const _Float16* __restrict__ ph,
                            const float* __restrict__ pb,
                            float* __restrict__ out)
{
  const int lane = threadIdx.x & 31;
  const int wave = threadIdx.x >> 5;
  const int m0   = blockIdx.x * 64 + wave * 16;
  const int f0   = blockIdx.y * 64;
  const int kb2  = (lane < 16) ? 0 : 8;
  const int koff = (lane < 16) ? 0 : 16;

  const _Float16* arow = att + (size_t)(m0 + (lane & 15)) * CDIM;

  int ofeat[4];
  const _Float16* brow[4];
#pragma unroll
  for (int nt = 0; nt < 4; ++nt) {
    ofeat[nt] = f0 + nt * 16 + (lane & 15);
    brow[nt]  = ph + (size_t)ofeat[nt] * CDIM + koff;
  }

  v16h a = load_a_frag(arow, kb2);
  v16h bf[4];
#pragma unroll
  for (int nt = 0; nt < 4; ++nt) bf[nt] = *(const v16h*)(brow[nt]);

  v8f acc[4] = {};
#pragma unroll 2
  for (int kt = 0; kt < 12; ++kt) {
    const int ktn = (kt < 11) ? kt + 1 : kt;
    v16h an = load_a_frag(arow + ktn * 32, kb2);
    v16h bn[4];
#pragma unroll
    for (int nt = 0; nt < 4; ++nt) bn[nt] = *(const v16h*)(brow[nt] + ktn * 32);

#pragma unroll
    for (int nt = 0; nt < 4; ++nt) acc[nt] = wmma_f16(a, bf[nt], acc[nt]);

    a = an;
#pragma unroll
    for (int nt = 0; nt < 4; ++nt) bf[nt] = bn[nt];
  }

#pragma unroll
  for (int nt = 0; nt < 4; ++nt) {
    float bias = pb[ofeat[nt]];
#pragma unroll
    for (int r = 0; r < 8; ++r) {
      int row = (lane < 16) ? r : 8 + r;
      out[(size_t)(m0 + row) * CDIM + ofeat[nt]] = acc[nt][r] + bias;
    }
  }
}

// ---------------------------------------------------------------------------
extern "C" void kernel_launch(void* const* d_in, const int* in_sizes, int n_in,
                              void* d_out, int out_size, void* d_ws, size_t ws_size,
                              hipStream_t stream)
{
  const float* x      = (const float*)d_in[0];
  const float* q_w    = (const float*)d_in[1];
  const float* q_b    = (const float*)d_in[2];
  const float* kv_w   = (const float*)d_in[3];
  const float* kv_b   = (const float*)d_in[4];
  const float* proj_w = (const float*)d_in[5];
  const float* proj_b = (const float*)d_in[6];
  float* out = (float*)d_out;

  char* ws = (char*)d_ws;
  const size_t qp_bytes  = (size_t)BATCH * NH * SEQ * HDP * sizeof(_Float16); // 18.9 MB
  const size_t v_bytes   = (size_t)BATCH * NH * SEQ * HD  * sizeof(_Float16); // 14.2 MB
  const size_t att_bytes = (size_t)BATCH * SEQ * CDIM * sizeof(_Float16);     // 14.2 MB
  const size_t xh_bytes  = (size_t)BATCH * SEQ * CDIM * sizeof(_Float16);     // 14.2 MB
  const size_t wh_bytes  = (size_t)3 * CDIM * CDIM * sizeof(_Float16);        // 0.9 MB
  size_t off = 0;
  _Float16* qp  = (_Float16*)(ws + off); off += qp_bytes;
  _Float16* kp  = (_Float16*)(ws + off); off += qp_bytes;
  _Float16* vT  = (_Float16*)(ws + off); off += v_bytes;
  _Float16* att = (_Float16*)(ws + off); off += att_bytes;
  _Float16* xh  = (_Float16*)(ws + off); off += xh_bytes;
  _Float16* wh  = (_Float16*)(ws + off); off += wh_bytes;   // q_w | kv_w
  _Float16* phw = (_Float16*)(ws + off); off += (size_t)CDIM * CDIM * sizeof(_Float16);

  // Zero q/k so the hd 48..63 padding contributes nothing to QK^T.
  hipMemsetAsync(qp, 0, qp_bytes, stream);
  hipMemsetAsync(kp, 0, qp_bytes, stream);

  dim3 blk(128), cblk(256);
  const int n4x  = BATCH * SEQ * CDIM / 4;     // 1,769,472
  const int n4q  = CDIM * CDIM / 4;            // 36,864
  const int n4kv = 2 * CDIM * CDIM / 4;        // 73,728
  cvt_kernel<<<dim3((n4x + 255) / 256),  cblk, 0, stream>>>(x,      xh,  n4x);
  cvt_kernel<<<dim3((n4q + 255) / 256),  cblk, 0, stream>>>(q_w,    wh,  n4q);
  cvt_kernel<<<dim3((n4kv + 255) / 256), cblk, 0, stream>>>(kv_w,   wh + (size_t)CDIM * CDIM, n4kv);
  cvt_kernel<<<dim3((n4q + 255) / 256),  cblk, 0, stream>>>(proj_w, phw, n4q);

  dim3 g1((BATCH * SEQ) / 64, (3 * CDIM) / 64);   // 288 x 18
  qkv_kernel<<<g1, blk, 0, stream>>>(xh, wh, q_b, kv_b, qp, kp, vT);

  dim3 g2(BATCH * NH, SEQ / 64);                  // 64 x 36
  attn_kernel<<<g2, blk, 0, stream>>>(qp, kp, vT, att);

  dim3 g3((BATCH * SEQ) / 64, CDIM / 64);         // 288 x 6
  proj_kernel<<<g3, blk, 0, stream>>>(att, phw, proj_b, out);
}
